// hr_pose_generator_19963007991839
// MI455X (gfx1250) — matrix operand order repaired and checked by reading, not compiled, not run on hardware
//
#include <hip/hip_runtime.h>
#include <math.h>

// MI455X (gfx1250) implementation.
// ~350 GFLOP dense work vs ~0.3GB data -> compute bound: route all large GEMMs
// through v_wmma_f32_16x16x32_f16 (f16 operands, f32 accum). Weights (~15MB f16)
// sit in the 192MB L2. Per-wave 32x64 WMMA tiles (8 wmma / K-step, A reused over
// 4 N-tiles) give ~21 FLOP per L2 byte so the matrix pipes, not L2, are the
// limiter. Attention (128->512->128 MLP over 640k rows) is fused per (b,i) with
// an LDS-held T matrix and chained WMMA GEMMs.

typedef __attribute__((ext_vector_type(16))) _Float16 v16h;
typedef __attribute__((ext_vector_type(8)))  _Float16 v8h;
typedef __attribute__((ext_vector_type(8)))  float    v8f;

#define MODE_BIAS        0
#define MODE_BNRELU      1
#define MODE_BNRELU_RES  2
#define MODE_TANH        3

// ---- WMMA fragment loaders (contiguous k = 16*(L>>4) + s, same order for A and B) ----
// A operand (16x32 f16, row-major source, ld in elements):
// lane L: row = row0 + (L&15); 16 contiguous K values at k0 + (L>>4)*16
__device__ __forceinline__ v16h ldfragA(const _Float16* __restrict__ base, int ld,
                                        int row0, int k0, int lane) {
  int r  = row0 + (lane & 15);
  int ko = (lane >> 4) << 4;
  const _Float16* p = base + (size_t)r * ld + k0 + ko;
  v8h lo = *reinterpret_cast<const v8h*>(p);
  v8h hi = *reinterpret_cast<const v8h*>(p + 8);
  return __builtin_shufflevector(lo, hi, 0,1,2,3,4,5,6,7,8,9,10,11,12,13,14,15);
}
// B operand (32x16 f16). B[k,n] = W[n,k] (W row-major N x K). Lane L holds
// column n = row0 + (L&15), 16 contiguous K values at k0 + (L>>4)*16.
__device__ __forceinline__ v16h ldfragB(const _Float16* __restrict__ base, int ld,
                                        int row0, int k0, int lane) {
  int r  = row0 + (lane & 15);
  int ko = (lane >> 4) << 4;
  return *reinterpret_cast<const v16h*>(base + (size_t)r * ld + k0 + ko);
}

// ---- generic WMMA GEMM: D(M x N) = epilogue(A(MxK,f16) @ W(NxK,f16)^T) ----
// one wave per 32x64 output tile; grid = (N/64, M/32); requires M%32==0, N%64==0
__global__ __launch_bounds__(32) void k_gemm(
    const _Float16* __restrict__ A, const _Float16* __restrict__ W,
    const float* __restrict__ bias, int nbias,
    const float* __restrict__ scale, const float* __restrict__ shift,
    const float* __restrict__ resid,
    float* __restrict__ outF, int outLd, int outCols,
    _Float16* __restrict__ outH,
    int N, int K, int mode)
{
  int lane = threadIdx.x;
  int n0 = blockIdx.x * 64;
  int m0 = blockIdx.y * 32;
  v8f zero = {};
  v8f acc[2][4];
#pragma unroll
  for (int mt = 0; mt < 2; ++mt)
#pragma unroll
    for (int nt = 0; nt < 4; ++nt) acc[mt][nt] = zero;

  for (int kk = 0; kk < K; kk += 32) {
    v16h a0 = ldfragA(A, K, m0,      kk, lane);
    v16h a1 = ldfragA(A, K, m0 + 16, kk, lane);
    v16h b0 = ldfragB(W, K, n0,      kk, lane);
    v16h b1 = ldfragB(W, K, n0 + 16, kk, lane);
    v16h b2 = ldfragB(W, K, n0 + 32, kk, lane);
    v16h b3 = ldfragB(W, K, n0 + 48, kk, lane);
    if (kk + 32 < K) {  // gfx1250 global_prefetch_b8 of next K-slice
      __builtin_prefetch(A + (size_t)(m0 + (lane & 15)) * K + kk + 32, 0, 1);
      __builtin_prefetch(W + (size_t)(n0 + (lane & 15)) * K + kk + 32, 0, 1);
    }
    acc[0][0] = __builtin_amdgcn_wmma_f32_16x16x32_f16(false, a0, false, b0, (short)0, acc[0][0], false, false);
    acc[0][1] = __builtin_amdgcn_wmma_f32_16x16x32_f16(false, a0, false, b1, (short)0, acc[0][1], false, false);
    acc[0][2] = __builtin_amdgcn_wmma_f32_16x16x32_f16(false, a0, false, b2, (short)0, acc[0][2], false, false);
    acc[0][3] = __builtin_amdgcn_wmma_f32_16x16x32_f16(false, a0, false, b3, (short)0, acc[0][3], false, false);
    acc[1][0] = __builtin_amdgcn_wmma_f32_16x16x32_f16(false, a1, false, b0, (short)0, acc[1][0], false, false);
    acc[1][1] = __builtin_amdgcn_wmma_f32_16x16x32_f16(false, a1, false, b1, (short)0, acc[1][1], false, false);
    acc[1][2] = __builtin_amdgcn_wmma_f32_16x16x32_f16(false, a1, false, b2, (short)0, acc[1][2], false, false);
    acc[1][3] = __builtin_amdgcn_wmma_f32_16x16x32_f16(false, a1, false, b3, (short)0, acc[1][3], false, false);
  }

  int nl = lane & 15;
  int mh = (lane >> 4) << 3;
#pragma unroll
  for (int nt = 0; nt < 4; ++nt) {
    int n = n0 + nt * 16 + nl;
    float bv = 0.f, sc = 1.f, sh = 0.f;
    if (mode == MODE_BIAS || mode == MODE_TANH) {
      if (bias && n < nbias) bv = bias[n];
    } else { sc = scale[n]; sh = shift[n]; }
#pragma unroll
    for (int mt = 0; mt < 2; ++mt) {
#pragma unroll
      for (int r = 0; r < 8; ++r) {
        int m = m0 + mt * 16 + r + mh;
        float v = acc[mt][nt][r];
        if (mode == MODE_BIAS)      v += bv;
        else if (mode == MODE_TANH) v = tanhf(v + bv);
        else {
          v = fmaxf(sc * v + sh, 0.f);
          if (mode == MODE_BNRELU_RES) v += resid[(size_t)m * N + n];
        }
        if (outF && n < outCols) outF[(size_t)m * outLd + n] = v;
        if (outH)                outH[(size_t)m * N + n] = (_Float16)v;
      }
    }
  }
}

// ---- GRU (10->10, 50 steps), one thread per batch row; trivial FLOPs ----
__global__ __launch_bounds__(256) void k_gru(
    const float* __restrict__ noise, const float* __restrict__ Wih,
    const float* __restrict__ Whh, const float* __restrict__ bih,
    const float* __restrict__ bhh, float* __restrict__ aux)
{
  int b = threadIdx.x;
  float h[10];
#pragma unroll
  for (int d = 0; d < 10; ++d) h[d] = 0.f;
  for (int t = 0; t < 50; ++t) {
    const float* x = noise + ((size_t)b * 50 + t) * 10;
    float xr[10];
#pragma unroll
    for (int d = 0; d < 10; ++d) xr[d] = x[d];
    float gi[30], gh[30];
    for (int g = 0; g < 30; ++g) {
      float si = bih[g], s2 = bhh[g];
#pragma unroll
      for (int d = 0; d < 10; ++d) { si += Wih[g*10+d] * xr[d]; s2 += Whh[g*10+d] * h[d]; }
      gi[g] = si; gh[g] = s2;
    }
    float hn[10];
#pragma unroll
    for (int d = 0; d < 10; ++d) {
      float r = 1.f / (1.f + __expf(-(gi[d]    + gh[d])));
      float z = 1.f / (1.f + __expf(-(gi[10+d] + gh[10+d])));
      float n = tanhf(gi[20+d] + r * gh[20+d]);
      hn[d] = (1.f - z) * n + z * h[d];
    }
#pragma unroll
    for (int d = 0; d < 10; ++d) {
      h[d] = hn[d];
      aux[((size_t)b * 50 + t) * 10 + d] = tanhf(hn[d]);
    }
  }
}

// ---- pack x267 = [input(256) | aux(10) | sk(1)] -> f16, K padded to 288 ----
__global__ void k_pack(const float* __restrict__ input, const float* __restrict__ aux,
                       const float* __restrict__ fsk, const float* __restrict__ fW,
                       const float* __restrict__ fb, _Float16* __restrict__ X)
{
  int row = blockIdx.x;          // b*50+t
  int b = row / 50, t = row % 50;
  for (int c = threadIdx.x; c < 288; c += blockDim.x) {
    float v;
    if (c < 256)      v = input[(size_t)row * 256 + c];
    else if (c < 266) v = aux[(size_t)row * 10 + (c - 256)];
    else if (c == 266) {
      float s = fb[t];
      for (int k2 = 0; k2 < 36; ++k2) s += fsk[(size_t)b * 36 + k2] * fW[t * 36 + k2];
      v = s;
    } else v = 0.f;
    X[(size_t)row * 288 + c] = (_Float16)v;
  }
}

// ---- f32 -> f16 weight convert with zero padding ----
__global__ void k_cvt(const float* __restrict__ src, _Float16* __restrict__ dst,
                      int rows, int cols, int dstLd, int dstRows)
{
  int idx = blockIdx.x * blockDim.x + threadIdx.x;
  int total = dstRows * dstLd;
  if (idx >= total) return;
  int r = idx / dstLd, c = idx % dstLd;
  float v = (r < rows && c < cols) ? src[(size_t)r * cols + c] : 0.f;
  dst[idx] = (_Float16)v;
}

// ---- fold Linear bias + BN(eval) into scale/shift: y = relu(scale*acc+shift) ----
__global__ void k_bnprep(const float* __restrict__ g, const float* __restrict__ b,
                         const float* __restrict__ m, const float* __restrict__ v,
                         const float* __restrict__ rb,
                         float* __restrict__ scale, float* __restrict__ shift, int n)
{
  int i = blockIdx.x * blockDim.x + threadIdx.x;
  if (i >= n) return;
  float s = g[i] * rsqrtf(v[i] + 1e-5f);
  scale[i] = s;
  shift[i] = s * (rb[i] - m[i]) + b[i];
}

// ---- fused point-transformer attention: one WG (4 waves) per (b,i) ----
#define ATT_SMEM (32768 + 12800 + 512 + 4096)
__global__ __launch_bounds__(128) void k_attn(
    const float* __restrict__ qkv,    // 12800 x 384 (q|k|v)
    const float* __restrict__ posp,   // 12800 x 32 (xyz in cols 0..2)
    const _Float16* __restrict__ AW1, const float* __restrict__ ab1,  // 512x128, 512
    const _Float16* __restrict__ AW2, const float* __restrict__ ab2,  // 128x512, 128
    const float* __restrict__ pW1, const float* __restrict__ pb1,     // 64x3, 64
    const float* __restrict__ pW2, const float* __restrict__ pb2,     // 128x64, 128
    _Float16* __restrict__ aggH)      // 12800 x 128
{
  __shared__ __align__(16) char smem[ATT_SMEM];
  _Float16* Tsh  = (_Float16*)smem;                    // 64x128 f16 (phase B input)
  float*    Ssh  = (float*)smem;                       // 64x128 f32 (aliases Tsh, phase C)
  _Float16* relH = (_Float16*)(smem + 32768);          // 50x128 f16
  float*    qish = (float*)(smem + 32768 + 12800);     // 128 f32
  _Float16* Hs   = (_Float16*)(smem + 32768 + 12800 + 512); // 4 waves x 16x32 f16

  int tid  = threadIdx.x;
  int rowI = blockIdx.x;            // b*50 + i
  int b    = rowI / 50;

  if (tid < 128) qish[tid] = qkv[(size_t)rowI * 384 + tid];
  __syncthreads();

  float pix = posp[(size_t)rowI * 32 + 0];
  float piy = posp[(size_t)rowI * 32 + 1];
  float piz = posp[(size_t)rowI * 32 + 2];

  // phase A: rel_emb MLP (3->64->128) + T = q_i - k_j + rel (VALU, tiny)
  if (tid < 50) {
    int j = tid;
    size_t rowJ = (size_t)b * 50 + j;
    float dx = pix - posp[rowJ * 32 + 0];
    float dy = piy - posp[rowJ * 32 + 1];
    float dz = piz - posp[rowJ * 32 + 2];
    float hid[64];
#pragma unroll
    for (int hh = 0; hh < 64; ++hh) {
      float u = pW1[hh*3+0]*dx + pW1[hh*3+1]*dy + pW1[hh*3+2]*dz + pb1[hh];
      hid[hh] = fmaxf(u, 0.f);
    }
    for (int d = 0; d < 128; ++d) {
      float re = pb2[d];
#pragma unroll
      for (int hh = 0; hh < 64; ++hh) re += pW2[d*64+hh] * hid[hh];
      relH[j*128 + d] = (_Float16)re;
      float kv = qkv[rowJ * 384 + 128 + d];
      Tsh[j*128 + d] = (_Float16)(qish[d] - kv + re);
    }
  }
  for (int idx = tid; idx < 14 * 128; idx += 128) Tsh[50*128 + idx] = (_Float16)0.f;
  __syncthreads();

  // phase B: per wave m-tile of 16 j-rows; S = (relu(T@W1^T+b1))@W2^T, K=512 streamed
  int wave = tid >> 5, lane = tid & 31;
  int m0 = wave * 16;
  v8f zero = {};
  v8f S[8];
#pragma unroll
  for (int t = 0; t < 8; ++t) S[t] = zero;
  _Float16* hsw = Hs + wave * 512;
  int nl = lane & 15, mh = (lane >> 4) << 3;

  for (int hb = 0; hb < 16; ++hb) {
    v8f h0 = {}; v8f h1 = {};
#pragma unroll
    for (int kk = 0; kk < 128; kk += 32) {
      v16h a  = ldfragA(Tsh, 128, m0, kk, lane);
      v16h b0 = ldfragB(AW1, 128, hb*32,      kk, lane);
      v16h b1 = ldfragB(AW1, 128, hb*32 + 16, kk, lane);
      h0 = __builtin_amdgcn_wmma_f32_16x16x32_f16(false, a, false, b0, (short)0, h0, false, false);
      h1 = __builtin_amdgcn_wmma_f32_16x16x32_f16(false, a, false, b1, (short)0, h1, false, false);
    }
    float bb0 = ab1[hb*32 + nl];
    float bb1 = ab1[hb*32 + 16 + nl];
#pragma unroll
    for (int r = 0; r < 8; ++r) {   // D-layout -> LDS (relu+f16), per-wave scratch
      int m = r + mh;
      hsw[m*32 + nl]      = (_Float16)fmaxf(h0[r] + bb0, 0.f);
      hsw[m*32 + 16 + nl] = (_Float16)fmaxf(h1[r] + bb1, 0.f);
    }
    v16h a2 = ldfragA(hsw, 32, 0, 0, lane);  // re-read in A layout (same wave)
#pragma unroll
    for (int nt = 0; nt < 8; ++nt) {
      v16h bw = ldfragB(AW2, 512, nt*16, hb*32, lane);
      S[nt] = __builtin_amdgcn_wmma_f32_16x16x32_f16(false, a2, false, bw, (short)0, S[nt], false, false);
    }
  }
  __syncthreads();   // everyone done reading Tsh before aliasing as Ssh
#pragma unroll
  for (int nt = 0; nt < 8; ++nt) {
    float bb = ab2[nt*16 + nl];
#pragma unroll
    for (int r = 0; r < 8; ++r) {
      int m = m0 + r + mh;
      Ssh[m*128 + nt*16 + nl] = S[nt][r] + bb;
    }
  }
  __syncthreads();

  // phase C: per-channel softmax over j (axis=-2) + weighted sum with v_ij
  int d = tid;
  float mx = -3.4e38f;
  for (int j = 0; j < 50; ++j) mx = fmaxf(mx, Ssh[j*128 + d]);
  float Z = 0.f, acc = 0.f;
  for (int j = 0; j < 50; ++j) {
    float e = __expf(Ssh[j*128 + d] - mx);
    Z += e;
    float vv = qkv[((size_t)b*50 + j) * 384 + 256 + d] + (float)relH[j*128 + d];
    acc += e * vv;
  }
  aggH[(size_t)rowI * 128 + d] = (_Float16)(acc / Z);
}

// ------------------------------ host side ------------------------------
extern "C" void kernel_launch(void* const* d_in, const int* in_sizes, int n_in,
                              void* d_out, int out_size, void* d_ws, size_t ws_size,
                              hipStream_t stream)
{
  const float* input    = (const float*)d_in[0];
  const float* first_sk = (const float*)d_in[1];
  const float* noise    = (const float*)d_in[2];
  const float* first_W  = (const float*)d_in[3];
  const float* first_b  = (const float*)d_in[4];
  const float* gru_Wih  = (const float*)d_in[5];
  const float* gru_Whh  = (const float*)d_in[6];
  const float* gru_bih  = (const float*)d_in[7];
  const float* gru_bhh  = (const float*)d_in[8];
  const float* l0_W     = (const float*)d_in[9];
  const float* l0_b     = (const float*)d_in[10];
  const float* res_W    = (const float*)d_in[11];
  const float* res_b    = (const float*)d_in[12];
  const float* bn_g     = (const float*)d_in[13];
  const float* bn_b     = (const float*)d_in[14];
  const float* bn_m     = (const float*)d_in[15];
  const float* bn_v     = (const float*)d_in[16];
  const float* fl1_W    = (const float*)d_in[17];
  const float* fl1_b    = (const float*)d_in[18];
  const float* fl2_W    = (const float*)d_in[19];
  const float* fl2_b    = (const float*)d_in[20];
  const float* fl3_W    = (const float*)d_in[21];
  const float* fl3_b    = (const float*)d_in[22];
  const float* qkv_W    = (const float*)d_in[23];
  const float* posm_W1  = (const float*)d_in[24];
  const float* posm_b1  = (const float*)d_in[25];
  const float* posm_W2  = (const float*)d_in[26];
  const float* posm_b2  = (const float*)d_in[27];
  const float* attn_W1  = (const float*)d_in[28];
  const float* attn_b1  = (const float*)d_in[29];
  const float* attn_W2  = (const float*)d_in[30];
  const float* attn_b2  = (const float*)d_in[31];
  (void)in_sizes; (void)n_in; (void)out_size;

  const int M = 12800;   // B*50
  char* ws = (char*)d_ws;
  size_t off = 0;
  auto alloc = [&](size_t bytes) -> char* {
    char* p = ws + off;
    off += (bytes + 255) & ~(size_t)255;
    return p;
  };
  float*    OF32  = (float*)   alloc((size_t)M * 1024 * 4);
  _Float16* F0    = (_Float16*)alloc((size_t)M * 1024 * 2);
  _Float16* F1    = (_Float16*)alloc((size_t)M * 1024 * 2);
  _Float16* Y16   = (_Float16*)alloc((size_t)M * 1024 * 2);
  _Float16* XH    = (_Float16*)alloc((size_t)M * 288 * 2);
  _Float16* W0H   = (_Float16*)alloc((size_t)1024 * 288 * 2);
  _Float16* RWH   = (_Float16*)alloc((size_t)6144 * 1024 * 2);
  _Float16* FL1H  = (_Float16*)alloc((size_t)128 * 1024 * 2);
  _Float16* FL2H  = (_Float16*)alloc((size_t)64 * 1024 * 2);
  _Float16* QKVWH = (_Float16*)alloc((size_t)384 * 128 * 2);
  _Float16* AW1H  = (_Float16*)alloc((size_t)512 * 128 * 2);
  _Float16* AW2H  = (_Float16*)alloc((size_t)128 * 512 * 2);
  _Float16* FL3H  = (_Float16*)alloc((size_t)64 * 128 * 2);
  _Float16* FEATH = (_Float16*)alloc((size_t)M * 128 * 2);
  float*    QKV32 = (float*)   alloc((size_t)M * 384 * 4);
  float*    POSP  = (float*)   alloc((size_t)M * 32 * 4);
  _Float16* AGGH  = (_Float16*)alloc((size_t)M * 128 * 2);
  float*    AUX   = (float*)   alloc((size_t)M * 10 * 4);
  float*    SCALE = (float*)   alloc((size_t)6144 * 4);
  float*    SHIFT = (float*)   alloc((size_t)6144 * 4);
  (void)ws_size;  // requires ~181 MB of workspace

  // --- tiny sequential/setup kernels ---
  k_gru<<<1, 256, 0, stream>>>(noise, gru_Wih, gru_Whh, gru_bih, gru_bhh, AUX);
  k_pack<<<M, 128, 0, stream>>>(input, AUX, first_sk, first_W, first_b, XH);

  auto cvt = [&](const float* s, _Float16* d, int rows, int cols, int ld, int drows) {
    int total = drows * ld;
    k_cvt<<<(total + 255) / 256, 256, 0, stream>>>(s, d, rows, cols, ld, drows);
  };
  cvt(l0_W,    W0H,   1024, 267, 288, 1024);
  cvt(res_W,   RWH,   6144, 1024, 1024, 6144);
  cvt(fl1_W,   FL1H,  128, 1024, 1024, 128);
  cvt(fl2_W,   FL2H,  3,   1024, 1024, 64);
  cvt(qkv_W,   QKVWH, 384, 128, 128, 384);
  cvt(attn_W1, AW1H,  512, 128, 128, 512);
  cvt(attn_W2, AW2H,  128, 512, 512, 128);
  cvt(fl3_W,   FL3H,  36,  128, 128, 64);
  k_bnprep<<<24, 256, 0, stream>>>(bn_g, bn_b, bn_m, bn_v, res_b, SCALE, SHIFT, 6144);

  const int MB = M / 32;   // 400 row-blocks of 32

  // --- l0: (12800x288) @ (1024x288)^T + bias -> OF32 + F0 ---
  k_gemm<<<dim3(1024/64, MB), 32, 0, stream>>>(
      XH, W0H, l0_b, 1024, nullptr, nullptr, nullptr,
      OF32, 1024, 1024, F0, 1024, 288, MODE_BIAS);

  // --- 3 residual blocks (ping-pong f16 activations) ---
  _Float16* fin = F0;
  _Float16* fout = F1;
  for (int i = 0; i < 3; ++i) {
    const _Float16* Wa = RWH + (size_t)(i*2 + 0) * 1024 * 1024;
    const _Float16* Wb = RWH + (size_t)(i*2 + 1) * 1024 * 1024;
    k_gemm<<<dim3(1024/64, MB), 32, 0, stream>>>(
        fin, Wa, nullptr, 0, SCALE + (i*2)*1024, SHIFT + (i*2)*1024, nullptr,
        nullptr, 0, 0, Y16, 1024, 1024, MODE_BNRELU);
    k_gemm<<<dim3(1024/64, MB), 32, 0, stream>>>(
        Y16, Wb, nullptr, 0, SCALE + (i*2+1)*1024, SHIFT + (i*2+1)*1024, OF32,
        OF32, 1024, 1024, fout, 1024, 1024, MODE_BNRELU_RES);
    _Float16* tmp = fin; fin = fout; fout = tmp;
  }
  _Float16* trunk = fin;  // after 3 swaps: F1 holds final trunk (f16)

  // --- fl1 -> feats (f16), fl2 -> pos (N padded to 64), qkv (no bias) ---
  k_gemm<<<dim3(128/64, MB), 32, 0, stream>>>(
      trunk, FL1H, fl1_b, 128, nullptr, nullptr, nullptr,
      nullptr, 0, 0, FEATH, 128, 1024, MODE_BIAS);
  k_gemm<<<dim3(1, MB), 32, 0, stream>>>(
      trunk, FL2H, fl2_b, 3, nullptr, nullptr, nullptr,
      POSP, 32, 32, nullptr, 64, 1024, MODE_BIAS);
  k_gemm<<<dim3(384/64, MB), 32, 0, stream>>>(
      FEATH, QKVWH, nullptr, 0, nullptr, nullptr, nullptr,
      QKV32, 384, 384, nullptr, 384, 128, MODE_BIAS);

  // --- fused point-transformer attention ---
  k_attn<<<M, 128, 0, stream>>>(QKV32, POSP, AW1H, attn_b1, AW2H, attn_b2,
                                posm_W1, posm_b1, posm_W2, posm_b2, AGGH);

  // --- fl3 + tanh -> d_out (N padded to 64, write cols < 36) ---
  k_gemm<<<dim3(1, MB), 32, 0, stream>>>(
      AGGH, FL3H, fl3_b, 36, nullptr, nullptr, nullptr,
      (float*)d_out, 36, 36, nullptr, 64, 128, MODE_TANH);
}
